// EdgePredictor_37022618091734
// MI455X (gfx1250) — matrix operand order repaired and checked by reading, not compiled, not run on hardware
//
#include <hip/hip_runtime.h>

#define N_NODES 128
#define DFEAT   64
#define HID     256
#define OUTD    6
#define PPAIRS  8128          // 128*127/2
#define PTILE   16
#define TILES_PER_B (PPAIRS / PTILE)   // 508

typedef __bf16 bhalf;
typedef __bf16 v8y  __attribute__((ext_vector_type(8)));
typedef __bf16 v16y __attribute__((ext_vector_type(16)));
typedef float  v8f  __attribute__((ext_vector_type(8)));

__device__ __forceinline__ bhalf tobf(float f) {
    unsigned u = __builtin_bit_cast(unsigned, f);
    u += 0x7FFFu + ((u >> 16) & 1u);            // round-to-nearest-even
    unsigned short h = (unsigned short)(u >> 16);
    return __builtin_bit_cast(bhalf, h);
}

__device__ __forceinline__ float elu1(float z) {
    return z > 0.f ? z : (__expf(z) - 1.f);
}
// elu(elu(z)): elu(z)>0 iff z>0, so fuse the negative branch
__device__ __forceinline__ float elu2(float z) {
    return z > 0.f ? z : (__expf(__expf(z) - 1.f) - 1.f);
}

__device__ __forceinline__ v16y frag_from(const bhalf* p0, const bhalf* p1) {
    v8y lo = *(const v8y*)p0;
    v8y hi = *(const v8y*)p1;
    return __builtin_shufflevector(lo, hi, 0,1,2,3,4,5,6,7,8,9,10,11,12,13,14,15);
}

// Transpose fp32 weight [K][Nsrc] -> bf16 [Npad][K] (N-major), zero-pad cols.
__global__ void cvt_wt(const float* __restrict__ w, bhalf* __restrict__ wt,
                       int K, int Nsrc, int Npad) {
    int idx = blockIdx.x * blockDim.x + threadIdx.x;
    if (idx >= Npad * K) return;
    int n = idx / K;
    int k = idx - n * K;
    float v = (n < Nsrc) ? w[(size_t)k * Nsrc + n] : 0.f;
    wt[idx] = tobf(v);
}

__global__ __launch_bounds__(128) void edge_mlp(
    const float* __restrict__ x,
    const bhalf* __restrict__ w0t, const float* __restrict__ b0,
    const bhalf* __restrict__ w1t, const float* __restrict__ b1,
    const bhalf* __restrict__ w2t, const float* __restrict__ b2,
    float* __restrict__ out)
{
    __shared__ __align__(16) bhalf sA [2][PTILE][2 * DFEAT];  //  8 KB
    __shared__ __align__(16) bhalf sH1[2][PTILE][HID];        // 16 KB
    __shared__ __align__(16) bhalf sH2[2][PTILE][HID];        // 16 KB
    __shared__ float sC2[2][PTILE][16];                       //  2 KB
    __shared__ int   sIdx[2][PTILE];

    const int tid  = threadIdx.x;
    const int wave = tid >> 5;
    const int lane = tid & 31;
    const int ln16 = lane & 15;
    const int lhi  = lane >> 4;

    const int bidx  = blockIdx.x / TILES_PER_B;
    const int tile  = blockIdx.x % TILES_PER_B;
    const int pbase = tile * PTILE;

    // decode triu_indices(128, k=1) for the 16 pairs of this tile
    if (tid < PTILE) {
        int rem = pbase + tid;
        int a = 0;
        while (rem >= (N_NODES - 1 - a)) { rem -= (N_NODES - 1 - a); ++a; }
        sIdx[0][tid] = a;
        sIdx[1][tid] = a + 1 + rem;
    }
    __syncthreads();

    // stage elu(elu(f)) as bf16; row-major [pair][128]; mtile1 = halves swapped
    const float* xbase = x + (size_t)bidx * N_NODES * DFEAT;
    for (int i = tid; i < PTILE * 2 * DFEAT; i += 128) {
        int j = i >> 7;        // pair within tile
        int c = i & 127;       // column 0..127
        int node = (c < DFEAT) ? sIdx[0][j] : sIdx[1][j];
        int cc   = (c < DFEAT) ? c : c - DFEAT;
        bhalf hv = tobf(elu2(xbase[node * DFEAT + cc]));
        sA[0][j][c] = hv;
        sA[1][j][(c + DFEAT) & 127] = hv;
    }
    __syncthreads();

    const int mt  = wave & 1;          // M-tile (orientation block)
    const int ntb = (wave >> 1) * 8;   // first of 8 N-tiles for this wave

    // ---------------- layer 0: [16x128] @ [128x256] ----------------
    {
        v8f acc[8] = {};
        #pragma unroll
        for (int ks = 0; ks < 4; ++ks) {
            const bhalf* arow = &sA[mt][ln16][0];
            int k0 = ks * 32 + lhi * 8;
            v16y af = frag_from(arow + k0, arow + k0 + 16);
            #pragma unroll
            for (int t = 0; t < 8; ++t) {
                const bhalf* wp = w0t + (size_t)((ntb + t) * 16 + ln16) * 128
                                      + ks * 32 + lhi * 8;
                v16y bf = frag_from(wp, wp + 16);
                acc[t] = __builtin_amdgcn_wmma_f32_16x16x32_bf16(
                            false, af, false, bf, (short)0, acc[t], false, false);
            }
        }
        #pragma unroll
        for (int t = 0; t < 8; ++t) {
            int ncol = (ntb + t) * 16 + ln16;
            float bias = b0[ncol];
            #pragma unroll
            for (int r = 0; r < 8; ++r)
                sH1[mt][r + lhi * 8][ncol] = tobf(elu1(acc[t][r] + bias));
        }
    }
    __syncthreads();

    // ---------------- layer 1: [16x256] @ [256x256] ----------------
    {
        v8f acc[8] = {};
        #pragma unroll
        for (int ks = 0; ks < 8; ++ks) {
            const bhalf* arow = &sH1[mt][ln16][0];
            int k0 = ks * 32 + lhi * 8;
            v16y af = frag_from(arow + k0, arow + k0 + 16);
            #pragma unroll
            for (int t = 0; t < 8; ++t) {
                const bhalf* wp = w1t + (size_t)((ntb + t) * 16 + ln16) * 256
                                      + ks * 32 + lhi * 8;
                v16y bf = frag_from(wp, wp + 16);
                acc[t] = __builtin_amdgcn_wmma_f32_16x16x32_bf16(
                            false, af, false, bf, (short)0, acc[t], false, false);
            }
        }
        #pragma unroll
        for (int t = 0; t < 8; ++t) {
            int ncol = (ntb + t) * 16 + ln16;
            float bias = b1[ncol];
            #pragma unroll
            for (int r = 0; r < 8; ++r)
                sH2[mt][r + lhi * 8][ncol] = tobf(elu1(acc[t][r] + bias));
        }
    }
    __syncthreads();

    // ---------------- layer 2: [16x256] @ [256x16pad] ----------------
    if (wave < 2) {
        const int m2 = wave;
        v8f acc = {};
        #pragma unroll
        for (int ks = 0; ks < 8; ++ks) {
            const bhalf* arow = &sH2[m2][ln16][0];
            int k0 = ks * 32 + lhi * 8;
            v16y af = frag_from(arow + k0, arow + k0 + 16);
            const bhalf* wp = w2t + (size_t)ln16 * 256 + ks * 32 + lhi * 8;
            v16y bf = frag_from(wp, wp + 16);
            acc = __builtin_amdgcn_wmma_f32_16x16x32_bf16(
                      false, af, false, bf, (short)0, acc, false, false);
        }
        #pragma unroll
        for (int r = 0; r < 8; ++r)
            sC2[m2][r + lhi * 8][ln16] = acc[r];
    }
    __syncthreads();

    // mean over the two orientations, add b2, store
    if (tid < PTILE * OUTD) {
        int j = tid / OUTD, o = tid - j * OUTD;
        float res = 0.5f * (sC2[0][j][o] + sC2[1][j][o]) + b2[o];
        out[((size_t)(bidx * PPAIRS + pbase + j)) * OUTD + o] = res;
    }
}

extern "C" void kernel_launch(void* const* d_in, const int* in_sizes, int n_in,
                              void* d_out, int out_size, void* d_ws, size_t ws_size,
                              hipStream_t stream) {
    const float* x  = (const float*)d_in[0];
    const float* W0 = (const float*)d_in[1];
    const float* b0 = (const float*)d_in[2];
    const float* W1 = (const float*)d_in[3];
    const float* b1 = (const float*)d_in[4];
    const float* W2 = (const float*)d_in[5];
    const float* b2 = (const float*)d_in[6];
    float* out = (float*)d_out;

    bhalf* w0t = (bhalf*)d_ws;             // [256][128]
    bhalf* w1t = w0t + 256 * 128;          // [256][256]
    bhalf* w2t = w1t + 256 * 256;          // [16][256] (cols 6..15 zero)

    cvt_wt<<<(256 * 128 + 255) / 256, 256, 0, stream>>>(W0, w0t, 128, 256, 256);
    cvt_wt<<<(256 * 256 + 255) / 256, 256, 0, stream>>>(W1, w1t, 256, 256, 256);
    cvt_wt<<<(16  * 256 + 255) / 256, 256, 0, stream>>>(W2, w2t, 256, 6, 16);

    int B = in_sizes[0] / (N_NODES * DFEAT);
    edge_mlp<<<B * TILES_PER_B, 128, 0, stream>>>(x, w0t, b0, w1t, b1, w2t, b2, out);
}
